// CVAE_61658550501650
// MI455X (gfx1250) — matrix operand order repaired and checked by reading, not compiled, not run on hardware
//
#include <hip/hip_runtime.h>
#include <hip/hip_bf16.h>

// ---------------------------------------------------------------------------
// CVAE forward, MI455X (gfx1250, wave32, WMMA).
// Bandwidth-bound (fc_w = 340MB f32 @ 23.3TB/s dominates) => single-pass f32
// weight reads with NT (streaming) cache hint so the 340MB sweep does not
// evict the 1.2MB activation matrix from L2, in-register f32->f16 conversion,
// v_wmma_f32_16x16x32_f16 with f32 accumulate. Group-select hoisted BEFORE
// the big fc GEMM (halves traffic vs. reference's compute-both-then-where).
// ---------------------------------------------------------------------------

typedef __attribute__((ext_vector_type(16))) _Float16 v16h;
typedef __attribute__((ext_vector_type(8)))  float    v8f;
typedef __attribute__((ext_vector_type(4)))  float    v4f;

#define F_LOG2PI 1.8378770664093453f

enum { EPI_NONE = 0, EPI_RELU = 1, EPI_SIGMOID = 2, EPI_EXP = 3 };

// cached load (activations: re-read across N tiles, keep in L2/WGP$)
__device__ __forceinline__ v4f ld4(const float* p) {
    return *reinterpret_cast<const v4f*>(p);
}
// non-temporal load (weights: read exactly once, stream past the caches)
__device__ __forceinline__ v4f ntld4(const float* p) {
    return __builtin_nontemporal_load(reinterpret_cast<const v4f*>(p));
}

template<int BASE>
__device__ __forceinline__ void cvt8(v16h& d, v4f a, v4f b) {
    d[BASE + 0] = (_Float16)a.x; d[BASE + 1] = (_Float16)a.y;
    d[BASE + 2] = (_Float16)a.z; d[BASE + 3] = (_Float16)a.w;
    d[BASE + 4] = (_Float16)b.x; d[BASE + 5] = (_Float16)b.y;
    d[BASE + 6] = (_Float16)b.z; d[BASE + 7] = (_Float16)b.w;
}

// C[32,N] = act( A[32,K] @ W[N,K]^T + bias[N] ), optionally layered via
// blockIdx.y with per-layer strides. One wave per 16-column tile; two 16x16
// f32 accumulators cover M=32. K must be a multiple of 32.
//
// WMMA f16 operand layouts (ISA 7.12.2, wave32):
//   A 16x32 : lane half h, row r=lane&15; elem j -> K = h*8 + (j&7) + (j>>3)*16
//   B 32x16 : col n = lane&15;            elem j -> K = h*8 + (j&7) + (j>>3)*16
//   C 16x16 : VGPR v, row = h*8 + v, col = lane&15
template<int EPI>
__global__ __launch_bounds__(256)
void gemm_bias_act(const float* __restrict__ A, const float* __restrict__ W,
                   const float* __restrict__ bias, float* __restrict__ C,
                   int K, int N,
                   long wStride, long bStride, long aStride, long oStride)
{
    const int lane = threadIdx.x & 31;
    const int wave = threadIdx.x >> 5;
    const int wavesPerBlock = blockDim.x >> 5;
    const int nTile = (blockIdx.x * wavesPerBlock + wave) * 16;
    if (nTile >= N) return;                       // wave-uniform: EXEC stays all-1s

    const int layer = blockIdx.y;
    A    += (long)layer * aStride;
    W    += (long)layer * wStride;
    bias += (long)layer * bStride;
    C    += (long)layer * oStride;

    const int half = lane >> 4;
    const int r    = lane & 15;

    const float* wRow  = W + (long)(nTile + r) * K + half * 8;
    const float* aRow0 = A + (long)r        * K + half * 8;
    const float* aRow1 = A + (long)(r + 16) * K + half * 8;

    v8f acc0 = {};
    v8f acc1 = {};

    for (int kb = 0; kb < K; kb += 32) {
        __builtin_prefetch(wRow + kb + 256, 0, 0);   // global_prefetch_b8 on W stream

        v4f w0  = ntld4(wRow + kb);       v4f w1  = ntld4(wRow + kb + 4);
        v4f w2  = ntld4(wRow + kb + 16);  v4f w3  = ntld4(wRow + kb + 20);
        v4f a00 = ld4(aRow0 + kb);        v4f a01 = ld4(aRow0 + kb + 4);
        v4f a02 = ld4(aRow0 + kb + 16);   v4f a03 = ld4(aRow0 + kb + 20);
        v4f a10 = ld4(aRow1 + kb);        v4f a11 = ld4(aRow1 + kb + 4);
        v4f a12 = ld4(aRow1 + kb + 16);   v4f a13 = ld4(aRow1 + kb + 20);

        v16h bfrag, afrag0, afrag1;
        cvt8<0>(bfrag,  w0,  w1);  cvt8<8>(bfrag,  w2,  w3);
        cvt8<0>(afrag0, a00, a01); cvt8<8>(afrag0, a02, a03);
        cvt8<0>(afrag1, a10, a11); cvt8<8>(afrag1, a12, a13);

        acc0 = __builtin_amdgcn_wmma_f32_16x16x32_f16(
                   false, afrag0, false, bfrag, (short)0, acc0, false, false);
        acc1 = __builtin_amdgcn_wmma_f32_16x16x32_f16(
                   false, afrag1, false, bfrag, (short)0, acc1, false, false);
    }

    const float bcol = bias[nTile + r];
#pragma unroll
    for (int v = 0; v < 8; ++v) {
        const int row = half * 8 + v;
        float o0 = acc0[v] + bcol;
        float o1 = acc1[v] + bcol;
        if (EPI == EPI_RELU)    { o0 = fmaxf(o0, 0.f);            o1 = fmaxf(o1, 0.f); }
        if (EPI == EPI_SIGMOID) { o0 = 1.f / (1.f + __expf(-o0)); o1 = 1.f / (1.f + __expf(-o1)); }
        if (EPI == EPI_EXP)     { o0 = __expf(o0);                o1 = __expf(o1); }
        C[(long)row        * N + nTile + r] = o0;
        C[(long)(row + 16) * N + nTile + r] = o1;
    }
}

// --- helpers -----------------------------------------------------------------

__device__ __forceinline__ unsigned pcg(unsigned v) {
    unsigned s = v * 747796405u + 2891336453u;
    unsigned t = ((s >> ((s >> 28u) + 4u)) ^ s) * 277803737u;
    return (t >> 22u) ^ t;
}

__device__ __forceinline__ float gauss_hash(unsigned idx) {
    unsigned a = pcg(idx);
    unsigned b = pcg(idx ^ 0x9E3779B9u);
    float u1 = ((float)a + 1.0f) * 2.3283064365386963e-10f;  // (0,1]
    float u2 = (float)b * 2.3283064365386963e-10f;
    return sqrtf(-2.0f * logf(u1)) * __cosf(6.28318530717958f * u2);
}

// select mu/logvar by group, reparameterize, emit z and log_q
__global__ void reparam_kernel(const float* __restrict__ c,
                               const float* __restrict__ mu0, const float* __restrict__ mu1,
                               const float* __restrict__ lv0, const float* __restrict__ lv1,
                               float* __restrict__ z_ws, float* __restrict__ z_out,
                               float* __restrict__ lq_out)
{
    int idx = blockIdx.x * blockDim.x + threadIdx.x;
    if (idx >= 32 * 96) return;
    int b = idx / 96;
    bool is0 = (c[b] == 0.0f);
    float mu = is0 ? mu0[idx] : mu1[idx];
    float lv = is0 ? lv0[idx] : lv1[idx];
    float sd  = __expf(0.5f * lv);
    float eps = gauss_hash((unsigned)idx);
    float z   = mu + eps * sd;
    z_ws[idx]  = z;
    z_out[idx] = z;
    lq_out[idx] = -0.5f * eps * eps - 0.5f * lv - 0.5f * F_LOG2PI;  // (z-mu)/sd == eps
}

// pre_fc[b, i*96+j] = b_sel[i*96+j] + sum_k Wsel[k] * g[k,b,i] * g[k,b,j]
// (group-selected BEFORE the big fc GEMM -> one fc pass instead of two)
__global__ void outer_mix_kernel(const float* __restrict__ c,
                                 const float* __restrict__ g0, const float* __restrict__ g1,
                                 const float* __restrict__ W0, const float* __restrict__ b0,
                                 const float* __restrict__ W1, const float* __restrict__ b1,
                                 float* __restrict__ pre)
{
    int idx = blockIdx.x * blockDim.x + threadIdx.x;
    if (idx >= 32 * 9216) return;
    int b   = idx / 9216;
    int rem = idx - b * 9216;
    int i   = rem / 96;
    int j   = rem - i * 96;
    bool is0 = (c[b] == 0.0f);
    const float* g  = is0 ? g0 : g1;
    const float* Wm = is0 ? W0 : W1;
    const float* bv = is0 ? b0 : b1;
    float s = bv[rem];
#pragma unroll
    for (int k = 0; k < 4; ++k)
        s += Wm[k] * g[k * 3072 + b * 96 + i] * g[k * 3072 + b * 96 + j];
    pre[idx] = s;
}

// y[b] = reg_b + sum_j z[b,j]*reg_w[j] + c[b]*reg_w[96]
__global__ void reg_kernel(const float* __restrict__ z, const float* __restrict__ c,
                           const float* __restrict__ rw, const float* __restrict__ rb,
                           float* __restrict__ y)
{
    int b = threadIdx.x;
    if (b >= 32) return;
    float s = rb[0];
    for (int j = 0; j < 96; ++j) s += z[b * 96 + j] * rw[j];
    s += c[b] * rw[96];
    y[b] = s;
}

// ---------------------------------------------------------------------------

extern "C" void kernel_launch(void* const* d_in, const int* in_sizes, int n_in,
                              void* d_out, int out_size, void* d_ws, size_t ws_size,
                              hipStream_t stream)
{
    const float* x      = (const float*)d_in[0];
    const float* c      = (const float*)d_in[1];
    const float* enc_w1 = (const float*)d_in[2];
    const float* enc_b1 = (const float*)d_in[3];
    const float* enc_w2 = (const float*)d_in[4];
    const float* enc_b2 = (const float*)d_in[5];
    const float* mu0_w  = (const float*)d_in[6];
    const float* mu0_b  = (const float*)d_in[7];
    const float* lv0_w  = (const float*)d_in[8];
    const float* lv0_b  = (const float*)d_in[9];
    const float* mu1_w  = (const float*)d_in[10];
    const float* mu1_b  = (const float*)d_in[11];
    const float* lv1_w  = (const float*)d_in[12];
    const float* lv1_b  = (const float*)d_in[13];
    const float* dec_w  = (const float*)d_in[14];
    const float* dec_b  = (const float*)d_in[15];
    const float* gc0_w  = (const float*)d_in[16];
    const float* gc0_b  = (const float*)d_in[17];
    const float* gc1_w  = (const float*)d_in[18];
    const float* gc1_b  = (const float*)d_in[19];
    const float* fc_w   = (const float*)d_in[20];
    const float* fc_b   = (const float*)d_in[21];
    const float* reg_w  = (const float*)d_in[22];
    const float* reg_b  = (const float*)d_in[23];
    const float* W0     = (const float*)d_in[24];
    const float* b0     = (const float*)d_in[25];
    const float* W1     = (const float*)d_in[26];
    const float* b1     = (const float*)d_in[27];

    float* ws  = (float*)d_ws;
    float* hA  = ws;              // [32,1024]
    float* hB  = ws + 32768;      // [32,512]
    float* mu0 = ws + 49152;      // [32,96] x4
    float* mu1 = ws + 52224;
    float* lv0 = ws + 55296;
    float* lv1 = ws + 58368;
    float* zb  = ws + 61440;      // [32,96]
    float* db  = ws + 64512;      // [4,32,96]
    float* g0  = ws + 76800;      // [4,32,96]
    float* g1  = ws + 89088;      // [4,32,96]
    float* pre = ws + 101376;     // [32,9216]

    float* xout  = (float*)d_out;        // 294912
    float* yout  = xout + 294912;        // 32
    float* zout  = yout + 32;            // 3072
    float* lqout = zout + 3072;          // 3072

    const dim3 blk(256);

    // encoder
    gemm_bias_act<EPI_RELU><<<dim3(8, 1),  blk, 0, stream>>>(x,  enc_w1, enc_b1, hA, 9216, 1024, 0, 0, 0, 0);
    gemm_bias_act<EPI_RELU><<<dim3(4, 1),  blk, 0, stream>>>(hA, enc_w2, enc_b2, hB, 1024, 512,  0, 0, 0, 0);

    // group heads (selected in reparam)
    gemm_bias_act<EPI_NONE><<<dim3(1, 1), blk, 0, stream>>>(hB, mu0_w, mu0_b, mu0, 512, 96, 0, 0, 0, 0);
    gemm_bias_act<EPI_NONE><<<dim3(1, 1), blk, 0, stream>>>(hB, mu1_w, mu1_b, mu1, 512, 96, 0, 0, 0, 0);
    gemm_bias_act<EPI_NONE><<<dim3(1, 1), blk, 0, stream>>>(hB, lv0_w, lv0_b, lv0, 512, 96, 0, 0, 0, 0);
    gemm_bias_act<EPI_NONE><<<dim3(1, 1), blk, 0, stream>>>(hB, lv1_w, lv1_b, lv1, 512, 96, 0, 0, 0, 0);

    reparam_kernel<<<dim3(12), dim3(256), 0, stream>>>(c, mu0, mu1, lv0, lv1, zb, zout, lqout);

    // decoder + graph-convs, layered over blockIdx.y (L=4)
    gemm_bias_act<EPI_SIGMOID><<<dim3(1, 4), blk, 0, stream>>>(zb, dec_w, dec_b, db, 96, 96, 9216, 96, 0,    3072);
    gemm_bias_act<EPI_SIGMOID><<<dim3(1, 4), blk, 0, stream>>>(db, gc0_w, gc0_b, g0, 96, 96, 9216, 96, 3072, 3072);
    gemm_bias_act<EPI_SIGMOID><<<dim3(1, 4), blk, 0, stream>>>(db, gc1_w, gc1_b, g1, 96, 96, 9216, 96, 3072, 3072);

    // mixing outer products, group-selected pre-fc vector
    outer_mix_kernel<<<dim3(1152), dim3(256), 0, stream>>>(c, g0, g1, W0, b0, W1, b1, pre);

    // the big one: [32,9216] @ [9216,9216]^T with exp epilogue (340MB NT weight stream)
    gemm_bias_act<EPI_EXP><<<dim3(72, 1), blk, 0, stream>>>(pre, fc_w, fc_b, xout, 9216, 9216, 0, 0, 0, 0);

    // regression head
    reg_kernel<<<dim3(1), dim3(32), 0, stream>>>(zb, c, reg_w, reg_b, yout);
}
